// Attention_45792941310784
// MI455X (gfx1250) — matrix operand order repaired and checked by reading, not compiled, not run on hardware
//
#include <hip/hip_runtime.h>

// ---------- types ----------
typedef __bf16 bf16_t;
typedef __bf16 v16bf __attribute__((ext_vector_type(16)));
typedef __bf16 v8bf  __attribute__((ext_vector_type(8)));
typedef float  v8f   __attribute__((ext_vector_type(8)));

#define H_HEADS 16
#define T_SEQ   2048
#define C_DIM   1024
#define B_BATCH 2
#define M_ROWS  (B_BATCH * T_SEQ)   // 4096

// ---------- helpers ----------
__device__ __forceinline__ bf16_t f2bf(float f) {
    unsigned u = __builtin_bit_cast(unsigned, f);
    unsigned r = u + 0x7FFFu + ((u >> 16) & 1u);   // round-to-nearest-even
    unsigned short hsh = (unsigned short)(r >> 16);
    return __builtin_bit_cast(bf16_t, hsh);
}

__device__ __forceinline__ v8f wmma_bf16(v16bf a, v16bf b, v8f c) {
    return __builtin_amdgcn_wmma_f32_16x16x32_bf16(
        /*neg_a=*/false, a, /*neg_b=*/false, b,
        /*c_mod=*/(short)0, c, /*reuse_a=*/false, /*reuse_b=*/false);
}

// A-fragment (16x32 bf16, row-major source): lanes 0-15 -> K 0..15,
// lanes 16-31 -> K 16..31. Two contiguous 16B loads per lane.
__device__ __forceinline__ v16bf load_afrag(const bf16_t* rowk0, int sel /*0 or 8*/) {
    v8bf lo = *(const v8bf*)(rowk0 + 2 * sel);
    v8bf hi = *(const v8bf*)(rowk0 + 2 * sel + 8);
    v16bf a;
#pragma unroll
    for (int i = 0; i < 8; ++i) { a[i] = lo[i]; a[8 + i] = hi[i]; }
    return a;
}

__device__ __forceinline__ float rmax16(float v) {
    v = fmaxf(v, __shfl_xor(v, 1, 32));
    v = fmaxf(v, __shfl_xor(v, 2, 32));
    v = fmaxf(v, __shfl_xor(v, 4, 32));
    v = fmaxf(v, __shfl_xor(v, 8, 32));
    return v;
}
__device__ __forceinline__ float rsum16(float v) {
    v += __shfl_xor(v, 1, 32);
    v += __shfl_xor(v, 2, 32);
    v += __shfl_xor(v, 4, 32);
    v += __shfl_xor(v, 8, 32);
    return v;
}

// ---------- kernel 1: f32 -> bf16 ----------
__global__ void cvt_bf16(const float* __restrict__ in, bf16_t* __restrict__ out, int n) {
    int i = blockIdx.x * blockDim.x + threadIdx.x;
    int stride = gridDim.x * blockDim.x;
    for (; i < n; i += stride) out[i] = f2bf(in[i]);
}

// ---------- kernel 2: pack W (K=1024 x N=1024) into B-fragment tiles ----------
// tile = (kt,nt): 32x16 block. packed[ tile*512 + lane*16 + e ]
//   = W[ kt*32 + (lane<16?0:16) + e ][ nt*16 + (lane&15) ]
__global__ void pack_w(const float* __restrict__ W, bf16_t* __restrict__ Wp) {
    int idx  = blockIdx.x * blockDim.x + threadIdx.x;     // 0 .. 1048575
    int tile = idx >> 9;
    int lane = (idx >> 4) & 31;
    int e    = idx & 15;
    int kt = tile >> 6, nt = tile & 63;                   // numNt = 64
    int k = kt * 32 + ((lane >> 4) << 4) + e;
    int n = nt * 16 + (lane & 15);
    Wp[idx] = f2bf(W[k * C_DIM + n]);
}

// ---------- kernel 3: QKV projection GEMM (M=4096, N=1024, K=1024) ----------
// block = 128 threads (4 waves); block tile 64(M) x 64(N); wave tile 16 x 64.
// Unroll-by-2 double buffering: the two fragment sets alternate roles, so no
// register copies are needed to rotate buffers.
__global__ __launch_bounds__(128) void qkv_gemm(
    const bf16_t* __restrict__ X, const bf16_t* __restrict__ Wp,
    const float* __restrict__ bias, bf16_t* __restrict__ Out, int vmode) {
    int lane = threadIdx.x & 31;
    int w    = threadIdx.x >> 5;
    int hl = lane >> 4, ln = lane & 15, sel = hl * 8;
    int mblk = blockIdx.x, nblk = blockIdx.y;
    int mrow = mblk * 64 + w * 16 + ln;

    v8f zero = {0.f,0.f,0.f,0.f,0.f,0.f,0.f,0.f};
    v8f acc[4] = {zero, zero, zero, zero};

    const bf16_t* arow  = X + (size_t)mrow * C_DIM;
    const bf16_t* wbase = Wp + (size_t)(nblk * 4) * 512 + lane * 16;

    v16bf a0 = load_afrag(arow, sel);
    v16bf b0[4];
#pragma unroll
    for (int j = 0; j < 4; ++j) b0[j] = *(const v16bf*)(wbase + (size_t)j * 512);

    for (int kt = 0; kt < 32; kt += 2) {
        __builtin_prefetch(arow + kt * 32 + 256, 0, 3);
        // load fragment set for kt+1 while computing with kt
        v16bf a1 = load_afrag(arow + (kt + 1) * 32, sel);
        v16bf b1[4];
#pragma unroll
        for (int j = 0; j < 4; ++j)
            b1[j] = *(const v16bf*)(wbase + ((size_t)((kt + 1) * 64 + j)) * 512);
#pragma unroll
        for (int j = 0; j < 4; ++j) acc[j] = wmma_bf16(a0, b0[j], acc[j]);

        // load fragment set for kt+2 while computing with kt+1
        if (kt + 2 < 32) {
            a0 = load_afrag(arow + (kt + 2) * 32, sel);
#pragma unroll
            for (int j = 0; j < 4; ++j)
                b0[j] = *(const v16bf*)(wbase + ((size_t)((kt + 2) * 64 + j)) * 512);
        }
#pragma unroll
        for (int j = 0; j < 4; ++j) acc[j] = wmma_bf16(a1, b1[j], acc[j]);
    }

#pragma unroll
    for (int j = 0; j < 4; ++j) {
        int n = nblk * 64 + j * 16 + ln;
        float bv = bias[n];
#pragma unroll
        for (int r = 0; r < 8; ++r) {
            float val = acc[j][r] + bv;
            int m = mblk * 64 + w * 16 + r + 8 * hl;
            if (!vmode) {
                Out[(size_t)m * C_DIM + n] = f2bf(val);
            } else {  // store V transposed: (B, H, Dh, T)
                int bi = m >> 11, t = m & (T_SEQ - 1);
                int hh = n >> 6, dh = n & 63;
                Out[(((size_t)(bi * H_HEADS + hh)) * 64 + dh) * T_SEQ + t] = f2bf(val);
            }
        }
    }
}

// ---------- kernel 4: flash-style causal attention ----------
// grid (T/64, B*H), block 128 (4 waves); wave: 16 queries x full Dh=64.
__global__ __launch_bounds__(128) void attn_kernel(
    const bf16_t* __restrict__ Q, const bf16_t* __restrict__ K,
    const bf16_t* __restrict__ Vt, bf16_t* __restrict__ O) {
    __shared__ __align__(16) unsigned short sP[4][16][32];

    int lane = threadIdx.x & 31;
    int w    = threadIdx.x >> 5;
    int hl = lane >> 4, ln = lane & 15, sel = hl * 8;
    int q0 = blockIdx.x * 64;
    int bh = blockIdx.y;
    int bi = bh >> 4, hh = bh & 15;
    int qbase = q0 + w * 16;
    int qrow  = qbase + ln;

    const bf16_t* qptr = Q + ((size_t)(bi * T_SEQ + qrow)) * C_DIM + hh * 64;
    v16bf qa0 = load_afrag(qptr, sel);
    v16bf qa1 = load_afrag(qptr + 32, sel);

    // base pointers for K (row-major) and V^T fragments
    const bf16_t* kbp = K + ((size_t)(bi * T_SEQ + ln)) * C_DIM + hh * 64 + hl * 16;
    const bf16_t* vbp = Vt + ((size_t)(bh * 64 + ln)) * T_SEQ + hl * 16;

    v8f zero = {0.f,0.f,0.f,0.f,0.f,0.f,0.f,0.f};
    v8f oacc[4] = {zero, zero, zero, zero};
    float mrun[8], lrun[8];
#pragma unroll
    for (int r = 0; r < 8; ++r) { mrun[r] = -3.0e38f; lrun[r] = 0.f; }

    const float sc = 0.125f;  // 1/sqrt(64)
    const int kkend = qbase + 16;

    // preload K fragments for kk = 0 (tiles: [tile0 k0=0, tile0 k0=32, tile1 k0=0, tile1 k0=32])
    v16bf kb[4];
    kb[0] = *(const v16bf*)(kbp);
    kb[1] = *(const v16bf*)(kbp + 32);
    kb[2] = *(const v16bf*)(kbp + 16 * C_DIM);
    kb[3] = *(const v16bf*)(kbp + 16 * C_DIM + 32);

    for (int kk = 0; kk < kkend; kk += 32) {
        // --- S = Q K^T for 32 keys (two 16-wide tiles) ---
        v8f s0 = zero, s1 = zero;
        s0 = wmma_bf16(qa0, kb[0], s0);
        s0 = wmma_bf16(qa1, kb[1], s0);
        s1 = wmma_bf16(qa0, kb[2], s1);
        s1 = wmma_bf16(qa1, kb[3], s1);

        // --- issue next-block K loads early (double buffer) ---
        int kkn = kk + 32;
        v16bf kbn[4] = {kb[0], kb[1], kb[2], kb[3]};
        if (kkn < kkend) {
            const bf16_t* kp = kbp + (size_t)kkn * C_DIM;
            kbn[0] = *(const v16bf*)(kp);
            kbn[1] = *(const v16bf*)(kp + 32);
            kbn[2] = *(const v16bf*)(kp + 16 * C_DIM);
            kbn[3] = *(const v16bf*)(kp + 16 * C_DIM + 32);
        }
        // --- issue current V^T fragment loads early (hide under softmax) ---
        v16bf vb[4];
#pragma unroll
        for (int j = 0; j < 4; ++j)
            vb[j] = *(const v16bf*)(vbp + (size_t)(j * 16) * T_SEQ + kk);

        // --- causal mask + online softmax ---
        float p0[8], p1[8];
#pragma unroll
        for (int r = 0; r < 8; ++r) {
            int qr = qbase + r + 8 * hl;
            float a0 = ((kk + ln)      <= qr) ? s0[r] * sc : -3.0e38f;
            float a1 = ((kk + 16 + ln) <= qr) ? s1[r] * sc : -3.0e38f;
            float tm = rmax16(fmaxf(a0, a1));
            float mnew  = fmaxf(mrun[r], tm);
            float alpha = __expf(mrun[r] - mnew);
            mrun[r] = mnew;
            float e0 = __expf(a0 - mnew);
            float e1 = __expf(a1 - mnew);
            lrun[r] = lrun[r] * alpha + rsum16(e0 + e1);
            oacc[0][r] *= alpha; oacc[1][r] *= alpha;
            oacc[2][r] *= alpha; oacc[3][r] *= alpha;
            p0[r] = e0; p1[r] = e1;
        }

        // --- D-layout -> A-layout via per-wave LDS staging ---
#pragma unroll
        for (int r = 0; r < 8; ++r) {
            int row = r + 8 * hl;
            sP[w][row][ln]      = __builtin_bit_cast(unsigned short, f2bf(p0[r]));
            sP[w][row][16 + ln] = __builtin_bit_cast(unsigned short, f2bf(p1[r]));
        }
        __builtin_amdgcn_wave_barrier();
        asm volatile("s_wait_dscnt 0" ::: "memory");  // LDS stores visible to wave
        const unsigned short* pr = &sP[w][ln][0];
        v8bf plo = *(const v8bf*)(pr + 2 * sel);
        v8bf phi = *(const v8bf*)(pr + 2 * sel + 8);
        v16bf pa;
#pragma unroll
        for (int i = 0; i < 8; ++i) { pa[i] = plo[i]; pa[8 + i] = phi[i]; }
        __builtin_amdgcn_wave_barrier();

        // --- O += P * V ---
#pragma unroll
        for (int j = 0; j < 4; ++j) oacc[j] = wmma_bf16(pa, vb[j], oacc[j]);

#pragma unroll
        for (int j = 0; j < 4; ++j) kb[j] = kbn[j];
    }

    // --- normalize and store O (row-major (B*T, 1024)) ---
#pragma unroll
    for (int j = 0; j < 4; ++j) {
#pragma unroll
        for (int r = 0; r < 8; ++r) {
            float val = oacc[j][r] / lrun[r];
            int row = qbase + r + 8 * hl;
            O[((size_t)(bi * T_SEQ + row)) * C_DIM + hh * 64 + j * 16 + ln] = f2bf(val);
        }
    }
}

// ---------- kernel 5: output projection + bias + LayerNorm ----------
// grid = 256 blocks, block 128; block computes 16 full rows of y then LN.
__global__ __launch_bounds__(128) void proj_ln(
    const bf16_t* __restrict__ Ob, const bf16_t* __restrict__ Wop,
    const float* __restrict__ bo, const float* __restrict__ gamma,
    const float* __restrict__ beta, float* __restrict__ out) {
    __shared__ float ys[16][C_DIM];     // 64 KB
    __shared__ float red[16][8][2];
    __shared__ float stat[16][2];

    int tid = threadIdx.x;
    int lane = tid & 31, w = tid >> 5;
    int hl = lane >> 4, ln = lane & 15, sel = hl * 8;
    int m0 = blockIdx.x * 16;

    v8f zero = {0.f,0.f,0.f,0.f,0.f,0.f,0.f,0.f};
    v8f acc[16];
#pragma unroll
    for (int j = 0; j < 16; ++j) acc[j] = zero;

    const bf16_t* arow  = Ob + (size_t)(m0 + ln) * C_DIM;
    const bf16_t* wbase = Wop + (size_t)(w * 16) * 512 + lane * 16;

    v16bf a0 = load_afrag(arow, sel);
    for (int kt = 0; kt < 32; kt += 2) {
        // A for kt+1 in the alternate buffer while computing with kt
        v16bf a1 = load_afrag(arow + (kt + 1) * 32, sel);
#pragma unroll
        for (int g = 0; g < 4; ++g) {
            v16bf b[4];
#pragma unroll
            for (int j = 0; j < 4; ++j)
                b[j] = *(const v16bf*)(wbase + ((size_t)(kt * 64 + g * 4 + j)) * 512);
#pragma unroll
            for (int j = 0; j < 4; ++j)
                acc[g * 4 + j] = wmma_bf16(a0, b[j], acc[g * 4 + j]);
        }
        if (kt + 2 < 32) a0 = load_afrag(arow + (kt + 2) * 32, sel);
#pragma unroll
        for (int g = 0; g < 4; ++g) {
            v16bf b[4];
#pragma unroll
            for (int j = 0; j < 4; ++j)
                b[j] = *(const v16bf*)(wbase + ((size_t)((kt + 1) * 64 + g * 4 + j)) * 512);
#pragma unroll
            for (int j = 0; j < 4; ++j)
                acc[g * 4 + j] = wmma_bf16(a1, b[j], acc[g * 4 + j]);
        }
    }
#pragma unroll
    for (int j = 0; j < 16; ++j) {
        int col = w * 256 + j * 16 + ln;
        float bias = bo[col];
#pragma unroll
        for (int r = 0; r < 8; ++r) ys[r + 8 * hl][col] = acc[j][r] + bias;
    }
    __syncthreads();

    // LayerNorm: 8 threads per row
    int row = tid >> 3, sub = tid & 7;
    float s = 0.f, s2 = 0.f;
    for (int c = sub; c < C_DIM; c += 8) { float v = ys[row][c]; s += v; s2 += v * v; }
    red[row][sub][0] = s; red[row][sub][1] = s2;
    __syncthreads();
    if (tid < 16) {
        float S = 0.f, S2 = 0.f;
        for (int i = 0; i < 8; ++i) { S += red[tid][i][0]; S2 += red[tid][i][1]; }
        float mu  = S / (float)C_DIM;
        float var = S2 / (float)C_DIM - mu * mu;
        stat[tid][0] = mu;
        stat[tid][1] = rsqrtf(var + 1e-5f);
    }
    __syncthreads();
    for (int idx = tid; idx < 16 * C_DIM; idx += 128) {
        int r = idx >> 10, c = idx & (C_DIM - 1);
        float v = (ys[r][c] - stat[r][0]) * stat[r][1] * gamma[c] + beta[c];
        out[(size_t)(m0 + r) * C_DIM + c] = v;
    }
}

// ---------- host launcher ----------
extern "C" void kernel_launch(void* const* d_in, const int* in_sizes, int n_in,
                              void* d_out, int out_size, void* d_ws, size_t ws_size,
                              hipStream_t stream) {
    (void)in_sizes; (void)n_in; (void)out_size; (void)ws_size;
    const float* x     = (const float*)d_in[0];
    const float* Wq    = (const float*)d_in[1];
    const float* bq    = (const float*)d_in[2];
    const float* Wk    = (const float*)d_in[3];
    const float* bk    = (const float*)d_in[4];
    const float* Wv    = (const float*)d_in[5];
    const float* bv    = (const float*)d_in[6];
    const float* Wo    = (const float*)d_in[7];
    const float* bo    = (const float*)d_in[8];
    const float* gamma = (const float*)d_in[9];
    const float* beta  = (const float*)d_in[10];

    char* ws = (char*)d_ws;
    const size_t MB = 1024ull * 1024ull;
    bf16_t* xb  = (bf16_t*)(ws + 0 * MB);    // 8 MB
    bf16_t* Wqp = (bf16_t*)(ws + 8 * MB);    // 2 MB
    bf16_t* Wkp = (bf16_t*)(ws + 10 * MB);
    bf16_t* Wvp = (bf16_t*)(ws + 12 * MB);
    bf16_t* Wop = (bf16_t*)(ws + 14 * MB);
    bf16_t* Qb  = (bf16_t*)(ws + 16 * MB);   // 8 MB
    bf16_t* Kb  = (bf16_t*)(ws + 24 * MB);   // 8 MB
    bf16_t* Vt  = (bf16_t*)(ws + 32 * MB);   // 8 MB, transposed (B,H,Dh,T)
    bf16_t* Ob  = (bf16_t*)(ws + 40 * MB);   // 8 MB

    cvt_bf16<<<4096, 256, 0, stream>>>(x, xb, M_ROWS * C_DIM);
    pack_w<<<4096, 256, 0, stream>>>(Wq, Wqp);
    pack_w<<<4096, 256, 0, stream>>>(Wk, Wkp);
    pack_w<<<4096, 256, 0, stream>>>(Wv, Wvp);
    pack_w<<<4096, 256, 0, stream>>>(Wo, Wop);

    dim3 ggrid(64, 16);
    qkv_gemm<<<ggrid, 128, 0, stream>>>(xb, Wqp, bq, Qb, 0);
    qkv_gemm<<<ggrid, 128, 0, stream>>>(xb, Wkp, bk, Kb, 0);
    qkv_gemm<<<ggrid, 128, 0, stream>>>(xb, Wvp, bv, Vt, 1);

    dim3 agrid(T_SEQ / 64, B_BATCH * H_HEADS);
    attn_kernel<<<agrid, 128, 0, stream>>>(Qb, Kb, Vt, Ob);

    proj_ln<<<256, 128, 0, stream>>>(Ob, Wop, bo, gamma, beta, (float*)d_out);
}
